// CausalSelfAttention_12584254178030
// MI455X (gfx1250) — compile-verified, matrix operand order checked by reading, not run on hardware
//
#include <hip/hip_runtime.h>

// ---------------------------------------------------------------------------
// CausalSelfAttention for MI455X (gfx1250, wave32, WMMA 16x16x32 f16->f32)
// B=4, T=2048, E=1024, H=16, D=64
// Compute-bound (~206 GFLOP vs ~150MB HBM traffic) -> everything through the
// WMMA pipe; f16 operands, f32 accumulation.  K tiles staged into LDS via the
// Tensor Data Mover (tensor_load_to_lds + s_wait_tensorcnt).
// ---------------------------------------------------------------------------

typedef __attribute__((ext_vector_type(16))) _Float16 v16h;
typedef __attribute__((ext_vector_type(8)))  _Float16 v8h;
typedef __attribute__((ext_vector_type(8)))  float    v8f;
typedef __attribute__((ext_vector_type(4)))  unsigned int u32x4;
typedef __attribute__((ext_vector_type(4)))  int          i32x4;
typedef __attribute__((ext_vector_type(8)))  int          i32x8;

#define BB 4
#define TT 2048
#define EE 1024
#define HH 16
#define DD 64
#define LDQKV (3 * EE)

__device__ __forceinline__ v8f wmma32(v16h a, v16h b, v8f c) {
    // D = A(16x32 f16) * B(32x16 f16) + C(16x16 f32)
    return __builtin_amdgcn_wmma_f32_16x16x32_f16(
        /*neg_a=*/false, a, /*neg_b=*/false, b,
        /*c_mod=*/(short)0, c, /*reuse_a=*/false, /*reuse_b=*/false);
}

// Load a 16x32 f16 fragment where each lane reads from a ROW of a row-major
// matrix (A-fragments; also B-fragments of C = X*W^T since B[k][n] = W[n][k];
// also B-fragments of P*V when V is stored transposed as Vt[d][t]).
// Lane l (r = l&15, hf = l>>4) holds row r with K-values
// {k0+8*hf .. +7} in halves 0..7 and {k0+8*hf+16 .. +7} in halves 8..15,
// per the CDNA5 16-bit A-matrix VGPR layout.  Works for global or LDS-derived
// generic pointers (lowers to global_load_b128 / ds_load_b128).
__device__ __forceinline__ v16h load_row_frag(const _Float16* base, int ld, int k0) {
    const int lane = threadIdx.x & 31;
    const int r  = lane & 15;
    const int hf = lane >> 4;
    const _Float16* p = base + (size_t)r * ld + k0 + 8 * hf;
    v8h lo = *(const v8h*)(p);
    v8h hi = *(const v8h*)(p + 16);
    v16h f;
#pragma unroll
    for (int j = 0; j < 8; ++j) { f[j] = lo[j]; f[8 + j] = hi[j]; }
    return f;
}

// ---------------------------------------------------------------------------
// fp32 -> f16 conversion (grid-stride)
// ---------------------------------------------------------------------------
__global__ void __launch_bounds__(256) cvt_f32_f16(const float* __restrict__ in,
                                                   _Float16* __restrict__ out, int n) {
    for (int i = blockIdx.x * blockDim.x + threadIdx.x; i < n;
         i += gridDim.x * blockDim.x)
        out[i] = (_Float16)in[i];
}

// ---------------------------------------------------------------------------
// Transpose V out of the qkv buffer: Vt[((b*H + h)*D + d)*T + t] =
// qkv[(b*T + t)*3E + 2E + h*D + d].  Memory-bound but tiny (32 MB traffic).
// ---------------------------------------------------------------------------
__global__ void __launch_bounds__(256) transpose_v(const _Float16* __restrict__ qkv,
                                                   _Float16* __restrict__ vt) {
    const int n = BB * HH * DD * TT;
    for (int i = blockIdx.x * blockDim.x + threadIdx.x; i < n;
         i += gridDim.x * blockDim.x) {
        const int t   = i & (TT - 1);
        const int d   = (i >> 11) & (DD - 1);
        const int h   = (i >> 17) & (HH - 1);
        const int b   = i >> 21;
        vt[i] = qkv[(size_t)(b * TT + t) * LDQKV + 2 * EE + h * DD + d];
    }
}

// ---------------------------------------------------------------------------
// GEMM: C[M,N] = X[M,K] * W[N,K]^T, f16 in, f32-accum WMMA.
// One wave computes a 64x64 tile: 16 WMMAs per k-step against 8 fragment
// loads (2x the FLOP/byte of a 32x32 tile).  ~220 VGPRs, fine under wave32's
// 1024-VGPR budget for a compute-bound kernel.
// ---------------------------------------------------------------------------
template <bool F32OUT>
__global__ void __launch_bounds__(256) gemm_xwT(const _Float16* __restrict__ X,
                                                const _Float16* __restrict__ W,
                                                float* __restrict__ Cf,
                                                _Float16* __restrict__ Ch,
                                                int M, int N, int K) {
    const int lane = threadIdx.x & 31;
    const int r  = lane & 15;
    const int hf = lane >> 4;
    const int wid = blockIdx.x * (blockDim.x >> 5) + (threadIdx.x >> 5);
    const int ntiles = N >> 6;
    const int tm = wid / ntiles;
    const int tn = wid % ntiles;
    if (tm * 64 >= M) return;               // wave-uniform exit (EXEC stays all-1)
    const int m0 = tm * 64;
    const int n0 = tn * 64;

    v8f c[4][4];
#pragma unroll
    for (int im = 0; im < 4; ++im)
#pragma unroll
        for (int in = 0; in < 4; ++in) c[im][in] = (v8f){};

    const _Float16* Xr[4];
    const _Float16* Wr[4];
#pragma unroll
    for (int t = 0; t < 4; ++t) {
        Xr[t] = X + (size_t)(m0 + 16 * t) * K;
        Wr[t] = W + (size_t)(n0 + 16 * t) * K;
    }

    for (int k0 = 0; k0 < K; k0 += 32) {
        v16h a[4], b[4];
#pragma unroll
        for (int t = 0; t < 4; ++t) a[t] = load_row_frag(Xr[t], K, k0);
#pragma unroll
        for (int t = 0; t < 4; ++t) b[t] = load_row_frag(Wr[t], K, k0);

        if (k0 + 64 < K) {                  // pull next-next k-step into cache
            __builtin_prefetch(Xr[0] + (size_t)r * K + k0 + 64, 0, 1);
            __builtin_prefetch(Wr[0] + (size_t)r * K + k0 + 64, 0, 1);
        }

#pragma unroll
        for (int im = 0; im < 4; ++im)
#pragma unroll
            for (int in = 0; in < 4; ++in)
                c[im][in] = wmma32(a[im], b[in], c[im][in]);
    }

#pragma unroll
    for (int im = 0; im < 4; ++im)
#pragma unroll
        for (int in = 0; in < 4; ++in)
#pragma unroll
            for (int i = 0; i < 8; ++i) {
                const int row = m0 + 16 * im + i + 8 * hf;
                const int col = n0 + 16 * in + r;
                if (F32OUT)
                    Cf[(size_t)row * N + col] = c[im][in][i];
                else
                    Ch[(size_t)row * N + col] = (_Float16)c[im][in][i];
            }
}

// ---------------------------------------------------------------------------
// Flash attention: one wave per (b, h, 16-row q-tile).  4 waves / block,
// disjoint per-wave LDS regions (no cross-wave sync needed).
// K tiles (32 x 64 f16, row stride 3E) are DMA'd into LDS by the Tensor Data
// Mover; V comes from the transposed vt[b][h][d][t] buffer so P*V B-fragments
// are contiguous b128 loads.  O written as f16 [B*T][E].
// ---------------------------------------------------------------------------
__global__ void __launch_bounds__(128) attn_flash(const _Float16* __restrict__ qkv,
                                                  const _Float16* __restrict__ vt,
                                                  _Float16* __restrict__ O) {
    __shared__ _Float16 kbuf[4][32 * 64];   // TDM destination: K tile per wave
    __shared__ _Float16 plds[4][16 * 32];   // P tile staging per wave
    const int lane = threadIdx.x & 31;
    const int wv   = threadIdx.x >> 5;
    const int r  = lane & 15;
    const int hf = lane >> 4;

    const int wid = blockIdx.x * 4 + wv;
    const int qt = wid & (TT / 16 - 1);          // 128 q-tiles
    const int h  = (wid >> 7) & (HH - 1);
    const int b  = wid >> 11;
    const int q0 = qt * 16;

    const _Float16* Qb  = qkv + (size_t)(b * TT + q0) * LDQKV + h * DD;
    const _Float16* Kb  = qkv + (size_t)(b * TT) * LDQKV + EE + h * DD;
    const _Float16* VtB = vt  + (size_t)(b * HH + h) * DD * TT;   // [d][t]

    // Q fragments (16 rows x 64 d) = two K=32 fragments
    const v16h qa0 = load_row_frag(Qb, LDQKV, 0);
    const v16h qa1 = load_row_frag(Qb, LDQKV, 32);

    v8f o[4];                                   // O accum: 16 rows x 64 d
#pragma unroll
    for (int nt = 0; nt < 4; ++nt) o[nt] = (v8f){};

    float mst[8], lst[8];                       // online-softmax row stats
#pragma unroll
    for (int i = 0; i < 8; ++i) { mst[i] = -3.0e38f; lst[i] = 0.0f; }

    const float scale = 0.125f;                 // 1/sqrt(64)

    // TDM descriptor pieces that don't change per iteration
    const unsigned lds_off = (unsigned)(size_t)(&kbuf[wv][0]);
    i32x8 g1;
    g1[0] = 0x00010000;          // workgroup_mask=0, data_size=1 (2 bytes)
    g1[1] = (int)(DD << 16);     // tensor_dim0 = 64       (bits  79:48 lo)
    g1[2] = (int)(TT << 16);     // tensor_dim1 = 2048     (bits 111:80 lo)
    g1[3] = (int)(DD << 16);     // tile_dim0   = 64       (bits 127:112)
    g1[4] = 32;                  // tile_dim1   = 32, tile_dim2 = 0
    g1[5] = LDQKV;               // tensor_dim0_stride = 3072 (bits 207:160 lo)
    g1[6] = 0;                   // dim0_stride hi, dim1_stride lo
    g1[7] = 0;                   // tensor_dim1_stride hi
    const i32x4 gz  = (i32x4)0;  // 2D tensor: groups 2/3 unused
    const i32x8 gz8 = (i32x8)0;

    for (int j0 = 0; j0 < q0 + 16; j0 += 32) {  // wave-uniform trip count
        // ---- TDM: DMA K tile rows [j0, j0+32) x 64 f16 into LDS ----------
        {
            const unsigned long long ga =
                (unsigned long long)(size_t)(Kb + (size_t)j0 * LDQKV);
            u32x4 g0;
            g0[0] = 1u;                               // count=1 (valid, user)
            g0[1] = lds_off;                          // lds_addr
            g0[2] = (unsigned)(ga & 0xFFFFFFFFu);     // global_addr[31:0]
            g0[3] = (unsigned)((ga >> 32) & 0x1FFFFFFu) | (2u << 30); // +type=2
            asm volatile("" ::: "memory");
            __builtin_amdgcn_tensor_load_to_lds(g0, g1, gz, gz, gz8, 0);
            __builtin_amdgcn_s_wait_tensorcnt(0);
            asm volatile("" ::: "memory");
        }

        // ---- S = scale * Q K^T for 32 keys (two 16x16 tiles) -------------
        v8f s[2];
#pragma unroll
        for (int jt = 0; jt < 2; ++jt) {
            const _Float16* Krow = &kbuf[wv][jt * 16 * DD];
            v16h kb0 = load_row_frag(Krow, DD, 0);
            v16h kb1 = load_row_frag(Krow, DD, 32);
            v8f acc = (v8f){};
            acc = wmma32(qa0, kb0, acc);
            acc = wmma32(qa1, kb1, acc);
#pragma unroll
            for (int i = 0; i < 8; ++i) {
                const int row = q0 + i + 8 * hf;
                const int col = j0 + jt * 16 + r;
                float v = acc[i] * scale;
                s[jt][i] = (col <= row) ? v : -3.0e38f;   // causal mask
            }
        }

        // ---- online softmax (row lives in one 16-lane half, one vgpr) ----
#pragma unroll
        for (int i = 0; i < 8; ++i) {
            float t = fmaxf(s[0][i], s[1][i]);
            t = fmaxf(t, __shfl_xor(t, 1, 32));
            t = fmaxf(t, __shfl_xor(t, 2, 32));
            t = fmaxf(t, __shfl_xor(t, 4, 32));
            t = fmaxf(t, __shfl_xor(t, 8, 32));
            const float mnew  = fmaxf(mst[i], t);
            const float alpha = __expf(mst[i] - mnew);
            const float p0 = __expf(s[0][i] - mnew);
            const float p1 = __expf(s[1][i] - mnew);
            s[0][i] = p0;
            s[1][i] = p1;
            float ls = p0 + p1;
            ls += __shfl_xor(ls, 1, 32);
            ls += __shfl_xor(ls, 2, 32);
            ls += __shfl_xor(ls, 4, 32);
            ls += __shfl_xor(ls, 8, 32);
            lst[i] = lst[i] * alpha + ls;
            mst[i] = mnew;
#pragma unroll
            for (int nt = 0; nt < 4; ++nt) o[nt][i] *= alpha;
        }

        // ---- re-layout P: C-fragment -> LDS -> A-fragment ----------------
        _Float16* lw = plds[wv];
#pragma unroll
        for (int i = 0; i < 8; ++i) {
            const int row = i + 8 * hf;
            lw[row * 32 + r]      = (_Float16)s[0][i];
            lw[row * 32 + 16 + r] = (_Float16)s[1][i];
        }
        // Same-wave LDS ordering; also guarantees all kbuf DS reads retired
        // before the next iteration's TDM overwrites the tile.
        asm volatile("s_wait_dscnt 0" ::: "memory");

        v16h pa;
        {
            const v8h lo = *(const v8h*)(lw + r * 32 + 8 * hf);
            const v8h hi = *(const v8h*)(lw + r * 32 + 8 * hf + 16);
#pragma unroll
            for (int j = 0; j < 8; ++j) { pa[j] = lo[j]; pa[8 + j] = hi[j]; }
        }

        // ---- O += P(16x32) * V(32x64) via Vt (contiguous b128 loads) -----
        // B[k][n] = V[j0+k][nt*16+n] = Vt[(nt*16+n)*T + j0+k]
#pragma unroll
        for (int nt = 0; nt < 4; ++nt) {
            v16h vb = load_row_frag(VtB + (size_t)(nt * 16) * TT + j0, TT, 0);
            o[nt] = wmma32(pa, vb, o[nt]);
        }
    }

    // ---- epilogue: normalize and store O as f16 --------------------------
    _Float16* Ob = O + (size_t)(b * TT + q0) * EE + h * DD;
#pragma unroll
    for (int i = 0; i < 8; ++i) {
        const float inv = 1.0f / lst[i];
        const int row = i + 8 * hf;
#pragma unroll
        for (int nt = 0; nt < 4; ++nt)
            Ob[(size_t)row * EE + nt * 16 + r] = (_Float16)(o[nt][i] * inv);
    }
}

// ---------------------------------------------------------------------------
// Host launch
// ---------------------------------------------------------------------------
extern "C" void kernel_launch(void* const* d_in, const int* in_sizes, int n_in,
                              void* d_out, int out_size, void* d_ws, size_t ws_size,
                              hipStream_t stream) {
    const float* x     = (const float*)d_in[0];   // [B,T,E]
    const float* W_qkv = (const float*)d_in[1];   // [3E,E]
    const float* W_out = (const float*)d_in[2];   // [E,E]
    float* out = (float*)d_out;                   // [B,T,E]

    const size_t MT = (size_t)BB * TT;            // 8192 rows

    _Float16* xh    = (_Float16*)d_ws;                      // 8192*1024
    _Float16* wqkvh = xh    + MT * EE;                      // 3072*1024
    _Float16* wouth = wqkvh + (size_t)3 * EE * EE;          // 1024*1024
    _Float16* qkvh  = wouth + (size_t)EE * EE;              // 8192*3072
    _Float16* oh    = qkvh  + MT * 3 * EE;                  // 8192*1024
    _Float16* vth   = oh    + MT * EE;                      // 8192*1024 (V^T)

    cvt_f32_f16<<<2048, 256, 0, stream>>>(x,     xh,    (int)(MT * EE));
    cvt_f32_f16<<<1024, 256, 0, stream>>>(W_qkv, wqkvh, 3 * EE * EE);
    cvt_f32_f16<<< 512, 256, 0, stream>>>(W_out, wouth, EE * EE);

    // QKV projection: [8192,1024] x [3072,1024]^T -> [8192,3072] (f16)
    {
        const int waves = (int)(MT / 64) * (3 * EE / 64);   // 128*48 = 6144
        gemm_xwT<false><<<waves / 8, 256, 0, stream>>>(
            xh, wqkvh, nullptr, qkvh, (int)MT, 3 * EE, EE);
    }

    // Transpose V for contiguous P*V fragment loads
    transpose_v<<<2048, 256, 0, stream>>>(qkvh, vth);

    // Flash attention: B*H*(T/16) = 8192 waves, 4 waves/block
    attn_flash<<<(BB * HH * (TT / 16)) / 4, 128, 0, stream>>>(qkvh, vth, oh);

    // Output projection: [8192,1024] x [1024,1024]^T -> [8192,1024] (f32)
    {
        const int waves = (int)(MT / 64) * (EE / 64);       // 128*16 = 2048
        gemm_xwT<true><<<waves / 8, 256, 0, stream>>>(
            oh, wouth, out, nullptr, (int)MT, EE, EE);
    }
}
